// TimeGraph_61521111548349
// MI455X (gfx1250) — compile-verified, matrix operand order checked by reading, not run on hardware
//
#include <hip/hip_runtime.h>

#define BATCH 8
#define TT 16
#define NNODES 2048
#define CH 512

typedef __attribute__((ext_vector_type(16))) __bf16 v16bf;
typedef __attribute__((ext_vector_type(8)))  __bf16 v8bf;
typedef __attribute__((ext_vector_type(8)))  float  v8f;

// Build a 16-bit WMMA A/B fragment from two contiguous 16-byte runs.
// Layout (CDNA5 ISA 7.12.2, 16-bit 16x32): lane l holds row/col (l&15);
// element e: K = (e>>3)*16 + (l>>4)*8 + (e&7)  -> two b128 loads.
static __device__ __forceinline__ v16bf make_frag(const __bf16* p0, const __bf16* p1) {
    v8bf lo = *(const v8bf*)p0;
    v8bf hi = *(const v8bf*)p1;
    return __builtin_shufflevector(lo, hi, 0,1,2,3,4,5,6,7,8,9,10,11,12,13,14,15);
}

static __device__ __forceinline__ v8f wmma_bf16(v16bf a, v16bf b, v8f c) {
    return __builtin_amdgcn_wmma_f32_16x16x32_bf16(false, a, false, b, (short)0, c, false, false);
}

// ---------------- conversion kernels ----------------
__global__ __launch_bounds__(256) void cvt_state_kernel(const float* __restrict__ nodes,
                                                        __bf16* __restrict__ st,
                                                        __bf16* __restrict__ stT) {
    size_t i = (size_t)blockIdx.x * 256 + threadIdx.x;
    if (i >= (size_t)BATCH * NNODES * CH) return;
    float v = nodes[i];
    __bf16 bv = (__bf16)v;
    st[i] = bv;
    size_t b   = i / ((size_t)NNODES * CH);
    size_t rem = i % ((size_t)NNODES * CH);
    size_t n = rem / CH, c = rem % CH;
    stT[(b * CH + c) * NNODES + n] = bv;
}

__global__ __launch_bounds__(256) void cvt_w_kernel(const float* __restrict__ w1,
                                                    const float* __restrict__ w2,
                                                    __bf16* __restrict__ w1b,
                                                    __bf16* __restrict__ w2b) {
    int i = blockIdx.x * 256 + threadIdx.x;
    if (i < CH * CH)            w1b[i]            = (__bf16)w1[i];
    else if (i < 2 * CH * CH)   w2b[i - CH * CH]  = (__bf16)w2[i - CH * CH];
}

// ---------------- scene mean + residual node mean ----------------
__global__ __launch_bounds__(256) void init_out_kernel(const float* __restrict__ vds,
                                                       const float* __restrict__ nodes,
                                                       float* __restrict__ out) {
    int idx = blockIdx.x * 256 + threadIdx.x;
    if (idx >= BATCH * CH) return;
    int b = idx / CH, c = idx % CH;
    float s = 0.f;
    for (int t = 0; t < TT; t++) s += vds[((size_t)b * TT + t) * CH + c];
    out[(size_t)b * 2 * CH + c] = s * (1.0f / TT);
    float g = 0.f;
    for (int n = 0; n < NNODES; n++) g += nodes[((size_t)b * NNODES + n) * CH + c];
    out[(size_t)b * 2 * CH + CH + c] = g * (1.0f / NNODES);   // residual part of gp_out
}

// ---------------- mat1 / mat2 = state @ W^T + b ----------------
// grid: (NNODES/16, CH/64, 2*BATCH), block 128 (4 waves), 1 16x16 tile / wave
__global__ __launch_bounds__(128) void gemm_lin_kernel(const __bf16* __restrict__ state,
                                                       const __bf16* __restrict__ w1b,
                                                       const __bf16* __restrict__ w2b,
                                                       const float* __restrict__ b1,
                                                       const float* __restrict__ b2,
                                                       __bf16* __restrict__ m1,
                                                       __bf16* __restrict__ m2) {
    int bz = blockIdx.z;
    int b = bz >> 1, which = bz & 1;
    const __bf16* W    = which ? w2b : w1b;
    const float*  bias = which ? b2  : b1;
    __bf16*       outm = which ? m2  : m1;

    int wave = threadIdx.x >> 5;
    int lane = threadIdx.x & 31;
    int h = lane >> 4, col = lane & 15;
    int i0 = blockIdx.x * 16;
    int n0 = blockIdx.y * 64 + wave * 16;

    const __bf16* Arow = state + ((size_t)b * NNODES + i0 + col) * CH;
    const __bf16* Brow = W + (size_t)(n0 + col) * CH;

    v8f acc = {};
    for (int k0 = 0; k0 < CH; k0 += 32) {
        v16bf a  = make_frag(Arow + k0 + h * 8, Arow + k0 + 16 + h * 8);
        v16bf bb = make_frag(Brow + k0 + h * 8, Brow + k0 + 16 + h * 8);
        acc = wmma_bf16(a, bb, acc);
    }
    float bv = bias[n0 + col];
    __bf16* obase = outm + (size_t)b * NNODES * CH;
    #pragma unroll
    for (int r = 0; r < 8; r++) {
        int m = i0 + r + 8 * h;              // D layout: VGPR r -> row r + 8*(lane>>4)
        obase[(size_t)m * CH + n0 + col] = (__bf16)(acc[r] + bv);
    }
}

// ---------------- fused adj -> softmax -> @state -> mean ----------------
// grid: (NNODES/16 row-blocks, BATCH), block 128 (4 waves).
// Cooperative: per 128-logit chunk, wave w computes logit columns
// [w*32, w*32+32); workgroup does joint online softmax in LDS; then each
// wave consumes the full K=128 P tile for its own 128 output channels.
__global__ __launch_bounds__(128) void attn_kernel(const __bf16* __restrict__ m1,
                                                   const __bf16* __restrict__ m2,
                                                   const __bf16* __restrict__ stT,
                                                   float* __restrict__ out) {
    __shared__ float  sS[16][128];                                   // logits (fp32)
    __shared__ __attribute__((aligned(16))) __bf16 sP[16][128];      // exp(S-m), bf16, K-fast
    __shared__ float  sPmax[16][8];
    __shared__ float  sPsum[16][8];
    __shared__ float  sM[16], sLrun[16], sScale[16];

    int b  = blockIdx.y;
    int i0 = blockIdx.x * 16;
    int wave = threadIdx.x >> 5;
    int lane = threadIdx.x & 31;
    int h = lane >> 4, col = lane & 15;
    int n0 = wave * 128;                 // this wave's output channel slice
    int jw = wave * 32;                  // this wave's logit columns within a chunk

    if (threadIdx.x < 16) { sM[threadIdx.x] = -1e30f; sLrun[threadIdx.x] = 0.f; }

    const __bf16* Arow = m1 + ((size_t)b * NNODES + i0 + col) * CH;

    // Hoist the mat1 A fragments (16 x v16bf = 128 VGPRs) for all logit GEMMs.
    v16bf afrag[16];
    #pragma unroll
    for (int ki = 0; ki < 16; ki++) {
        int k0 = ki * 32;
        afrag[ki] = make_frag(Arow + k0 + h * 8, Arow + k0 + 16 + h * 8);
    }

    v8f O[8];
    #pragma unroll
    for (int t = 0; t < 8; t++) O[t] = (v8f){};

    __syncthreads();   // sM/sLrun init visible

    for (int j0 = 0; j0 < NNODES; j0 += 128) {
        // ---- S quadrant: 16 x 32 logits (columns j0+jw .. j0+jw+31), K = 512 ----
        v8f s1 = {}, s2 = {};
        const __bf16* B1row = m2 + ((size_t)b * NNODES + j0 + jw + col) * CH;
        const __bf16* B2row = B1row + 16 * CH;
        if (j0 + 128 < NNODES)           // prefetch next chunk's mat2 rows (global_prefetch_b8)
            __builtin_prefetch(B1row + (size_t)128 * CH, 0, 1);
        for (int ki = 0; ki < 16; ki++) {
            int k0 = ki * 32;
            v16bf bf1 = make_frag(B1row + k0 + h * 8, B1row + k0 + 16 + h * 8);
            v16bf bf2 = make_frag(B2row + k0 + h * 8, B2row + k0 + 16 + h * 8);
            s1 = wmma_bf16(afrag[ki], bf1, s1);
            s2 = wmma_bf16(afrag[ki], bf2, s2);
        }
        #pragma unroll
        for (int r = 0; r < 8; r++) {
            sS[r + 8 * h][jw + col]      = s1[r];
            sS[r + 8 * h][jw + 16 + col] = s2[r];
        }
        __syncthreads();

        // ---- joint online softmax over the 16x128 chunk ----
        {   // partial max: thread t -> (row = t&15, 16-col segment = t>>4)
            int row = threadIdx.x & 15, seg = threadIdx.x >> 4;
            float pm = sS[row][seg * 16];
            for (int c = 1; c < 16; c++) pm = fmaxf(pm, sS[row][seg * 16 + c]);
            sPmax[row][seg] = pm;
        }
        __syncthreads();
        if (threadIdx.x < 16) {
            int row = threadIdx.x;
            float cmax = sPmax[row][0];
            for (int s = 1; s < 8; s++) cmax = fmaxf(cmax, sPmax[row][s]);
            float mo = sM[row];
            float mn = fmaxf(mo, cmax);
            sScale[row] = __expf(mo - mn);
            sM[row] = mn;
        }
        __syncthreads();
        {   // exp, bf16 P, partial sums
            int row = threadIdx.x & 15, seg = threadIdx.x >> 4;
            float mn = sM[row];
            float cs = 0.f;
            for (int c = 0; c < 16; c++) {
                float p = __expf(sS[row][seg * 16 + c] - mn);
                cs += p;
                sP[row][seg * 16 + c] = (__bf16)p;
            }
            sPsum[row][seg] = cs;
        }
        __syncthreads();
        if (threadIdx.x < 16) {
            int row = threadIdx.x;
            float cs = 0.f;
            for (int s = 0; s < 8; s++) cs += sPsum[row][s];
            sLrun[row] = sLrun[row] * sScale[row] + cs;
        }

        // ---- rescale running output, then O += P(16x128) @ state[j0:j0+128, n0:n0+128] ----
        float sc[8];
        #pragma unroll
        for (int r = 0; r < 8; r++) sc[r] = sScale[r + 8 * h];
        #pragma unroll
        for (int t = 0; t < 8; t++)
            #pragma unroll
            for (int r = 0; r < 8; r++) O[t][r] *= sc[r];

        for (int kk = 0; kk < 4; kk++) {
            v16bf pa = make_frag(&sP[col][kk * 32 + h * 8],
                                 &sP[col][kk * 32 + 16 + h * 8]);   // ds_load_b128 x2
            #pragma unroll
            for (int t = 0; t < 8; t++) {
                const __bf16* Bc = stT + ((size_t)b * CH + n0 + t * 16 + col) * NNODES
                                 + j0 + kk * 32;
                v16bf bf = make_frag(Bc + h * 8, Bc + 16 + h * 8);
                O[t] = wmma_bf16(pa, bf, O[t]);
            }
        }
        __syncthreads();   // all sS/sP readers done before next chunk overwrites
    }

    float li[8];
    #pragma unroll
    for (int r = 0; r < 8; r++) li[r] = sLrun[r + 8 * h];

    const float invN = 1.0f / (float)NNODES;
    #pragma unroll
    for (int t = 0; t < 8; t++) {
        float colsum = 0.f;
        #pragma unroll
        for (int r = 0; r < 8; r++) colsum += O[t][r] / li[r];
        atomicAdd(&out[(size_t)b * 2 * CH + CH + n0 + t * 16 + col], colsum * invN);
    }
}

// ---------------- launch ----------------
extern "C" void kernel_launch(void* const* d_in, const int* in_sizes, int n_in,
                              void* d_out, int out_size, void* d_ws, size_t ws_size,
                              hipStream_t stream) {
    const float* vd_s  = (const float*)d_in[0];   // [B,T,C]
    const float* nodes = (const float*)d_in[1];   // [B,N,C]
    const float* W1    = (const float*)d_in[2];   // [C,C]
    const float* b1    = (const float*)d_in[3];   // [C]
    const float* W2    = (const float*)d_in[4];   // [C,C]
    const float* b2    = (const float*)d_in[5];   // [C]
    float* out = (float*)d_out;

    char* ws = (char*)d_ws;
    size_t szState = (size_t)BATCH * NNODES * CH * sizeof(__bf16); // 16 MB
    size_t szW     = (size_t)CH * CH * sizeof(__bf16);             // 0.5 MB
    __bf16* stateBF  = (__bf16*)(ws);
    __bf16* stateTBF = (__bf16*)(ws + szState);
    __bf16* mat1BF   = (__bf16*)(ws + 2 * szState);
    __bf16* mat2BF   = (__bf16*)(ws + 3 * szState);
    __bf16* w1BF     = (__bf16*)(ws + 4 * szState);
    __bf16* w2BF     = (__bf16*)(ws + 4 * szState + szW);

    {   // bf16 conversions
        size_t total = (size_t)BATCH * NNODES * CH;
        cvt_state_kernel<<<(unsigned)((total + 255) / 256), 256, 0, stream>>>(nodes, stateBF, stateTBF);
        cvt_w_kernel<<<(2 * CH * CH + 255) / 256, 256, 0, stream>>>(W1, W2, w1BF, w2BF);
    }
    // scene mean + residual node mean (also makes gp slots deterministic each call)
    init_out_kernel<<<(BATCH * CH + 255) / 256, 256, 0, stream>>>(vd_s, nodes, out);

    // mat1 / mat2
    gemm_lin_kernel<<<dim3(NNODES / 16, CH / 64, 2 * BATCH), 128, 0, stream>>>(
        stateBF, w1BF, w2BF, b1, b2, mat1BF, mat2BF);

    // fused adj -> softmax -> @state -> mean(+atomic accumulate)
    attn_kernel<<<dim3(NNODES / 16, BATCH), 128, 0, stream>>>(mat1BF, mat2BF, stateTBF, out);

    (void)in_sizes; (void)n_in; (void)out_size; (void)ws_size;
}
